// Llama3_69544110457346
// MI455X (gfx1250) — matrix operand order repaired and checked by reading, not compiled, not run on hardware
//
#include <hip/hip_runtime.h>
#include <hip/hip_bf16.h>
#include <math.h>

// ---------------- model constants (match reference) ----------------
constexpr int Bc  = 2;
constexpr int Sc  = 1024;
constexpr int Dc  = 1024;
constexpr int Lc  = 8;
constexpr int HQc = 16;
constexpr int HKc = 4;
constexpr int DHc = 64;
constexpr int FFc = 2816;
constexpr int Vc  = 32000;
constexpr float EPSc   = 1e-5f;
constexpr float THETAc = 500000.0f;

typedef __attribute__((ext_vector_type(16))) _Float16 v16h;
typedef __attribute__((ext_vector_type(8)))  _Float16 v8h;
typedef __attribute__((ext_vector_type(4)))  _Float16 v4h;
typedef __attribute__((ext_vector_type(8)))  float    v8f;

enum { OP_NONE = 0, OP_SILU = 1, OP_ADD = 2 };

// =====================================================================
// Tiled GEMM: C[M,N] = A[M,K] * B[K,N]  (row-major f32 in memory).
// f32 -> f16 on the LDS staging path, f32 accumulate via
// v_wmma_f32_16x16x32_f16. Block tile 128x128, K-step 32, 8 waves
// (4M x 2N, each wave owns 2x4 16x16 tiles). Double-buffered LDS with
// register prefetch of the next K-tile: one barrier per K-step.
// Requires M%128==0, N%128==0, K%32==0 (true for every call here).
// OP is a compile-time template parameter -> straight-line epilogue.
// =====================================================================
template <int OP>
__launch_bounds__(256)
__global__ void gemm_wmma(const float* __restrict__ A,
                          const float* __restrict__ Bm,
                          const float* __restrict__ Res,
                          float* __restrict__ C,
                          int N, int K)
{
    // row stride 40 halfs = 80 B: rows stay 16B aligned, banks staggered
    __shared__ _Float16 lsA[2][128 * 40];   // A tile, [row][k]
    __shared__ _Float16 lsB[2][128 * 40];   // B tile transposed, [col][k]

    const int tid   = threadIdx.x;
    const int lane  = tid & 31;
    const int wave  = tid >> 5;
    const int waveM = wave & 3;             // 4 waves along M
    const int waveN = wave >> 2;            // 2 waves along N
    const int l16   = lane & 15;
    const int khalf = lane >> 4;            // 0 or 1 (half-wave)
    const int bRow  = blockIdx.y * 128;
    const int bCol  = blockIdx.x * 128;

    // per-thread staging coordinates: 4 float4 chunks for each tile
    // A tile 128x32: chunk i -> row rA, cols cA..cA+3
    // B tile 32x128: chunk i -> k kB, cols nB..nB+3 (stored transposed)
    int rA[4], cA[4], kB[4], nB[4];
#pragma unroll
    for (int i = 0; i < 4; ++i) {
        int idx4 = tid + i * 256;
        rA[i] = idx4 >> 3;
        cA[i] = (idx4 & 7) * 4;
        kB[i] = idx4 >> 5;
        nB[i] = (idx4 & 31) * 4;
    }

    v8f acc[2][4];
#pragma unroll
    for (int mi = 0; mi < 2; ++mi)
#pragma unroll
        for (int ni = 0; ni < 4; ++ni)
#pragma unroll
            for (int e = 0; e < 8; ++e) acc[mi][ni][e] = 0.0f;

    float4 ra[4], rb[4];

    // ---- prologue: fetch K-tile 0 and stage into buffer 0 ----
#pragma unroll
    for (int i = 0; i < 4; ++i) {
        ra[i] = *(const float4*)&A[(bRow + rA[i]) * K + cA[i]];
        rb[i] = *(const float4*)&Bm[kB[i] * N + bCol + nB[i]];
    }
#pragma unroll
    for (int i = 0; i < 4; ++i) {
        v4h ha = { (_Float16)ra[i].x, (_Float16)ra[i].y,
                   (_Float16)ra[i].z, (_Float16)ra[i].w };
        *(v4h*)&lsA[0][rA[i] * 40 + cA[i]] = ha;
        lsB[0][(nB[i] + 0) * 40 + kB[i]] = (_Float16)rb[i].x;
        lsB[0][(nB[i] + 1) * 40 + kB[i]] = (_Float16)rb[i].y;
        lsB[0][(nB[i] + 2) * 40 + kB[i]] = (_Float16)rb[i].z;
        lsB[0][(nB[i] + 3) * 40 + kB[i]] = (_Float16)rb[i].w;
    }
    __syncthreads();

    const int T = K >> 5;
    for (int t = 0; t < T; ++t) {
        const int cur = t & 1;
        const bool more = (t + 1) < T;

        // ---- prefetch next K-tile into registers (hides HBM latency) ----
        if (more) {
            const int k0 = (t + 1) * 32;
#pragma unroll
            for (int i = 0; i < 4; ++i) {
                ra[i] = *(const float4*)&A[(bRow + rA[i]) * K + k0 + cA[i]];
                rb[i] = *(const float4*)&Bm[(k0 + kB[i]) * N + bCol + nB[i]];
            }
        }

        // ---- build fragments per CDNA5 WMMA VGPR layout ----
        // A 16x32 f16: lane l (row=l%16): halfs 0..7 = K[kh*8..+7],
        //              halfs 8..15 = K[16+kh*8..+7]  (kh = l/16)
        v16h afr[2];
#pragma unroll
        for (int mi = 0; mi < 2; ++mi) {
            const _Float16* base = &lsA[cur][(waveM * 32 + mi * 16 + l16) * 40];
            v8h lo = *(const v8h*)(base + khalf * 8);
            v8h hi = *(const v8h*)(base + 16 + khalf * 8);
            afr[mi] = __builtin_shufflevector(lo, hi,
                0,1,2,3,4,5,6,7,8,9,10,11,12,13,14,15);
        }
        // B 32x16 f16: lane l (col=l%16): halfs 0..7 = K[kh*8..+7],
        //              halfs 8..15 = K[16+kh*8..+7]  (kh = l/16)
        v16h bfr[4];
#pragma unroll
        for (int ni = 0; ni < 4; ++ni) {
            const _Float16* base = &lsB[cur][(waveN * 64 + ni * 16 + l16) * 40];
            v8h lo = *(const v8h*)(base + khalf * 8);
            v8h hi = *(const v8h*)(base + 16 + khalf * 8);
            bfr[ni] = __builtin_shufflevector(lo, hi,
                0,1,2,3,4,5,6,7,8,9,10,11,12,13,14,15);
        }

#pragma unroll
        for (int mi = 0; mi < 2; ++mi)
#pragma unroll
            for (int ni = 0; ni < 4; ++ni)
                acc[mi][ni] = __builtin_amdgcn_wmma_f32_16x16x32_f16(
                    /*neg_a=*/false, afr[mi], /*neg_b=*/false, bfr[ni],
                    /*c_mod=*/(short)0, acc[mi][ni],
                    /*reuse_a=*/false, /*reuse_b=*/false);

        // ---- stage prefetched tile into the other buffer ----
        if (more) {
            const int nxt = cur ^ 1;
#pragma unroll
            for (int i = 0; i < 4; ++i) {
                v4h ha = { (_Float16)ra[i].x, (_Float16)ra[i].y,
                           (_Float16)ra[i].z, (_Float16)ra[i].w };
                *(v4h*)&lsA[nxt][rA[i] * 40 + cA[i]] = ha;
                lsB[nxt][(nB[i] + 0) * 40 + kB[i]] = (_Float16)rb[i].x;
                lsB[nxt][(nB[i] + 1) * 40 + kB[i]] = (_Float16)rb[i].y;
                lsB[nxt][(nB[i] + 2) * 40 + kB[i]] = (_Float16)rb[i].z;
                lsB[nxt][(nB[i] + 3) * 40 + kB[i]] = (_Float16)rb[i].w;
            }
            __syncthreads();
        }
    }

    // ---- epilogue: C/D layout: VGPR j -> row j + 8*(lane/16), col = lane%16
#pragma unroll
    for (int mi = 0; mi < 2; ++mi) {
#pragma unroll
        for (int ni = 0; ni < 4; ++ni) {
            const int row0 = bRow + waveM * 32 + mi * 16 + khalf * 8;
            const int col  = bCol + waveN * 64 + ni * 16 + l16;
            float* cp = C + row0 * N + col;
#pragma unroll
            for (int j = 0; j < 8; ++j) {
                float x = acc[mi][ni][j];
                if constexpr (OP == OP_SILU) {
                    x = x / (1.0f + __expf(-x));
                } else if constexpr (OP == OP_ADD) {
                    x += Res[row0 * N + col + j * N];
                }
                cp[j * N] = x;
            }
        }
    }
}

// =====================================================================
// Embedding gather: h[b,s,:] = tok_emb[tokens[b,s],:]
// =====================================================================
__global__ void embed_kernel(const int* __restrict__ tokens,
                             const float* __restrict__ emb,
                             float* __restrict__ h)
{
    long i = (long)blockIdx.x * blockDim.x + threadIdx.x;
    if (i >= (long)Bc * Sc * Dc) return;
    long bs = i / Dc;
    int  d  = (int)(i % Dc);
    h[i] = emb[(long)tokens[bs] * Dc + d];
}

// =====================================================================
// RMSNorm: one 256-thread block per row of length D=1024
// =====================================================================
__launch_bounds__(256)
__global__ void rmsnorm_kernel(const float* __restrict__ x,
                               const float* __restrict__ w,
                               float* __restrict__ out)
{
    __shared__ float red[256];
    const int row = blockIdx.x;
    const int tid = threadIdx.x;
    const float* xr = x + row * Dc;
    float s = 0.0f;
    for (int i = tid; i < Dc; i += 256) { float v = xr[i]; s += v * v; }
    red[tid] = s; __syncthreads();
    for (int st = 128; st > 0; st >>= 1) {
        if (tid < st) red[tid] += red[tid + st];
        __syncthreads();
    }
    float scale = rsqrtf(red[0] / (float)Dc + EPSc);
    float* op = out + row * Dc;
    for (int i = tid; i < Dc; i += 256) op[i] = xr[i] * scale * w[i];
}

// =====================================================================
// RoPE in-place on x laid out (B,S,H,DH); interleaved (even,odd) pairs
// =====================================================================
__global__ void rope_kernel(float* __restrict__ x, const int* __restrict__ start_pos, int H)
{
    int p = blockIdx.x * blockDim.x + threadIdx.x;
    int total = Bc * Sc * H * (DHc / 2);
    if (p >= total) return;
    int i = p & (DHc / 2 - 1);           // pair index 0..31
    int r = p >> 5;                      // (b,s,h) flat
    int h = r % H;
    int s = (r / H) % Sc;
    int b = r / (H * Sc);
    float freq = __expf(-((2.0f * (float)i) / (float)DHc) * __logf(THETAc));
    float t = (float)(start_pos[0] + s);
    float sn, cs;
    __sincosf(t * freq, &sn, &cs);
    int base = (((b * Sc + s) * H + h) * DHc) + 2 * i;
    float a  = x[base];
    float bb = x[base + 1];
    x[base]     = a * cs - bb * sn;
    x[base + 1] = a * sn + bb * cs;
}

// =====================================================================
// Causal attention with GQA (HQ=16 query heads share HK=4 kv heads).
// One block per (q_position, head, batch); scores staged in LDS.
// =====================================================================
__launch_bounds__(256)
__global__ void attn_kernel(const float* __restrict__ q,
                            const float* __restrict__ k,
                            const float* __restrict__ v,
                            float* __restrict__ o)
{
    __shared__ float sc[Sc];
    __shared__ float red[256];
    __shared__ float qs[DHc];
    const int qi  = blockIdx.x;
    const int h   = blockIdx.y;
    const int b   = blockIdx.z;
    const int tid = threadIdx.x;
    const int kh  = h / (HQc / HKc);

    const float* qrow = q + ((b * Sc + qi) * HQc + h) * DHc;
    if (tid < DHc) qs[tid] = qrow[tid];
    __syncthreads();

    // scores for k <= qi (causal)
    for (int kk = tid; kk <= qi; kk += 256) {
        const float* krow = k + ((b * Sc + kk) * HKc + kh) * DHc;
        float s = 0.0f;
#pragma unroll
        for (int d = 0; d < DHc; ++d) s += qs[d] * krow[d];
        sc[kk] = s * 0.125f;             // 1/sqrt(64)
    }
    __syncthreads();

    // block max
    float m = -1e30f;
    for (int kk = tid; kk <= qi; kk += 256) m = fmaxf(m, sc[kk]);
    red[tid] = m; __syncthreads();
    for (int st = 128; st > 0; st >>= 1) {
        if (tid < st) red[tid] = fmaxf(red[tid], red[tid + st]);
        __syncthreads();
    }
    m = red[0]; __syncthreads();

    // exp + block sum
    float lsum = 0.0f;
    for (int kk = tid; kk <= qi; kk += 256) {
        float p = __expf(sc[kk] - m);
        sc[kk] = p;
        lsum += p;
    }
    red[tid] = lsum; __syncthreads();
    for (int st = 128; st > 0; st >>= 1) {
        if (tid < st) red[tid] += red[tid + st];
        __syncthreads();
    }
    float inv = 1.0f / red[0];

    // weighted sum of V (one output dim per thread, coalesced V loads)
    if (tid < DHc) {
        float a = 0.0f;
        for (int kk = 0; kk <= qi; ++kk)
            a += sc[kk] * v[((b * Sc + kk) * HKc + kh) * DHc + tid];
        o[((b * Sc + qi) * HQc + h) * DHc + tid] = a * inv;
    }
}

// =====================================================================
// Elementwise gate multiply: a *= b
// =====================================================================
__global__ void mul_kernel(float* __restrict__ a, const float* __restrict__ b, long n)
{
    long i = (long)blockIdx.x * blockDim.x + threadIdx.x;
    if (i < n) a[i] *= b[i];
}

// =====================================================================
// Host-side orchestration (graph-capture safe: only kernel launches)
// =====================================================================
extern "C" void kernel_launch(void* const* d_in, const int* in_sizes, int n_in,
                              void* d_out, int out_size, void* d_ws, size_t ws_size,
                              hipStream_t stream)
{
    (void)in_sizes; (void)n_in; (void)out_size; (void)ws_size;

    const int*   tokens    = (const int*)  d_in[0];
    const int*   start_pos = (const int*)  d_in[1];
    const float* tok_emb   = (const float*)d_in[2];
    const float* wq        = (const float*)d_in[3];
    const float* wk        = (const float*)d_in[4];
    const float* wv        = (const float*)d_in[5];
    const float* wo        = (const float*)d_in[6];
    const float* w1        = (const float*)d_in[7];
    const float* w2        = (const float*)d_in[8];
    const float* w3        = (const float*)d_in[9];
    const float* anw       = (const float*)d_in[10];
    const float* fnw       = (const float*)d_in[11];
    const float* finw      = (const float*)d_in[12];
    const float* outw      = (const float*)d_in[13];
    float* out = (float*)d_out;

    const int M = Bc * Sc;               // 2048 token rows

    // workspace partition (floats); total ~21.0M floats (~83 MB)
    float* ws = (float*)d_ws;
    float* h  = ws; ws += (long)M * Dc;
    float* xn = ws; ws += (long)M * Dc;
    float* qb = ws; ws += (long)M * HQc * DHc;
    float* kb = ws; ws += (long)M * HKc * DHc;
    float* vb = ws; ws += (long)M * HKc * DHc;
    float* ao = ws; ws += (long)M * HQc * DHc;
    float* g1 = ws; ws += (long)M * FFc;
    float* g3 = ws; ws += (long)M * FFc;

    const dim3 blk(256);

    embed_kernel<<<(M * Dc + 255) / 256, blk, 0, stream>>>(tokens, tok_emb, h);

    for (int l = 0; l < Lc; ++l) {
        const float* wq_l = wq + (long)l * Dc * HQc * DHc;
        const float* wk_l = wk + (long)l * Dc * HKc * DHc;
        const float* wv_l = wv + (long)l * Dc * HKc * DHc;
        const float* wo_l = wo + (long)l * HQc * DHc * Dc;
        const float* w1_l = w1 + (long)l * Dc * FFc;
        const float* w2_l = w2 + (long)l * FFc * Dc;
        const float* w3_l = w3 + (long)l * Dc * FFc;

        // ---- attention block ----
        rmsnorm_kernel<<<M, blk, 0, stream>>>(h, anw + (long)l * Dc, xn);
        gemm_wmma<OP_NONE><<<dim3((HQc * DHc) / 128, M / 128), blk, 0, stream>>>(
            xn, wq_l, nullptr, qb, HQc * DHc, Dc);
        gemm_wmma<OP_NONE><<<dim3((HKc * DHc) / 128, M / 128), blk, 0, stream>>>(
            xn, wk_l, nullptr, kb, HKc * DHc, Dc);
        gemm_wmma<OP_NONE><<<dim3((HKc * DHc) / 128, M / 128), blk, 0, stream>>>(
            xn, wv_l, nullptr, vb, HKc * DHc, Dc);
        rope_kernel<<<(M * HQc * DHc / 2 + 255) / 256, blk, 0, stream>>>(qb, start_pos, HQc);
        rope_kernel<<<(M * HKc * DHc / 2 + 255) / 256, blk, 0, stream>>>(kb, start_pos, HKc);
        attn_kernel<<<dim3(Sc, HQc, Bc), blk, 0, stream>>>(qb, kb, vb, ao);
        gemm_wmma<OP_ADD><<<dim3(Dc / 128, M / 128), blk, 0, stream>>>(
            ao, wo_l, h, h, Dc, HQc * DHc);              // h += ao @ wo

        // ---- FFN block ----
        rmsnorm_kernel<<<M, blk, 0, stream>>>(h, fnw + (long)l * Dc, xn);
        gemm_wmma<OP_SILU><<<dim3(FFc / 128, M / 128), blk, 0, stream>>>(
            xn, w1_l, nullptr, g1, FFc, Dc);             // silu(x @ w1)
        gemm_wmma<OP_NONE><<<dim3(FFc / 128, M / 128), blk, 0, stream>>>(
            xn, w3_l, nullptr, g3, FFc, Dc);             // x @ w3
        mul_kernel<<<(int)(((long)M * FFc + 255) / 256), blk, 0, stream>>>(
            g1, g3, (long)M * FFc);
        gemm_wmma<OP_ADD><<<dim3(Dc / 128, M / 128), blk, 0, stream>>>(
            g1, w2_l, h, h, Dc, FFc);                    // h += gated @ w2
    }

    // ---- final norm + logits ----
    rmsnorm_kernel<<<M, blk, 0, stream>>>(h, finw, xn);
    gemm_wmma<OP_NONE><<<dim3(Vc / 128, M / 128), blk, 0, stream>>>(
        xn, outw, nullptr, out, Vc, Dc);
}
